// MultiHeadSelfAttention_10325101379799
// MI455X (gfx1250) — matrix operand order repaired and checked by reading, compiled, not run on hardware
//
#include <hip/hip_runtime.h>
#include <cstdint>

// ---------------------------------------------------------------------------
// MHA forward for MI455X (gfx1250): bf16 WMMA everywhere, flash attention.
// ---------------------------------------------------------------------------

#define DIM      768
#define NHEADS   12
#define HD       64
#define SEQ      2048
#define BATCH    2
#define NQKV     (3 * DIM)      // 2304
#define MROWS    (BATCH * SEQ)  // 4096

typedef __attribute__((ext_vector_type(16))) __bf16 v16bf;
typedef __attribute__((ext_vector_type(8)))  float  v8f;

__device__ __forceinline__ v8f zero8() {
    v8f z = {0.f, 0.f, 0.f, 0.f, 0.f, 0.f, 0.f, 0.f};
    return z;
}

// fp32 -> bf16 round-to-nearest-even, as raw uint16 bits
__device__ __forceinline__ uint16_t f2bf(float f) {
    uint32_t u = __float_as_uint(f);
    u += 0x7FFFu + ((u >> 16) & 1u);
    return (uint16_t)(u >> 16);
}

// A-fragment (16x32 bf16, row-major source): lane holds row (lane&15),
// K = koff..koff+15 contiguous,
// koff = (lane<16 ? 0 : 16). Caller passes p = rowPtr + koff.
__device__ __forceinline__ v16bf load_a_frag(const uint16_t* p) {
    v16bf a;
    uint4* d = reinterpret_cast<uint4*>(&a);
    d[0] = *reinterpret_cast<const uint4*>(p);
    d[1] = *reinterpret_cast<const uint4*>(p + 8);
    return a;
}

// B-fragment (32x16 bf16): lane holds col (lane&15), K = koff..koff+15
// contiguous, koff = (lane<16 ? 0 : 16). Caller passes p = colPtr + koff
// where the source is stored K-contiguous per column (i.e. transposed).
__device__ __forceinline__ v16bf load_b_frag(const uint16_t* p) {
    v16bf a;
    uint4* d = reinterpret_cast<uint4*>(&a);
    d[0] = *reinterpret_cast<const uint4*>(p);
    d[1] = *reinterpret_cast<const uint4*>(p + 8);
    return a;
}

__device__ __forceinline__ v8f wmma_bf16(v16bf a, v16bf b, v8f c) {
    return __builtin_amdgcn_wmma_f32_16x16x32_bf16(
        /*neg_a=*/false, a, /*neg_b=*/false, b,
        /*c_mod=*/(short)0, c, /*reuse_a=*/false, /*reuse_b=*/false);
}

// ---------------------------------------------------------------------------
// Kernel 1a: fp32 -> bf16 straight convert
// ---------------------------------------------------------------------------
__global__ void cvt_bf16_kernel(const float* __restrict__ in,
                                uint16_t* __restrict__ out, int n) {
    int i = blockIdx.x * blockDim.x + threadIdx.x;
    if (i < n) out[i] = f2bf(in[i]);
}

// ---------------------------------------------------------------------------
// Kernel 1b: fp32 [K,N] -> bf16 transposed [N,K]
// out[n*K + k] = in[k*N + n]
// ---------------------------------------------------------------------------
__global__ void cvt_transpose_kernel(const float* __restrict__ in,
                                     uint16_t* __restrict__ out,
                                     int K, int N) {
    int i = blockIdx.x * blockDim.x + threadIdx.x;
    if (i < K * N) {
        int n = i / K;
        int k = i % K;
        out[i] = f2bf(in[(size_t)k * N + n]);
    }
}

// ---------------------------------------------------------------------------
// Kernel 2: QKV GEMM.  C[4096,2304] = xb[4096,768] @ WqkvT^T + b
// One wave per 32(M)x64(N) macro-tile: 2 A-frags x 4 B-frags -> 8 WMMAs per
// 32-deep K step (24 B of operand loads per WMMA instead of 64 B).
// A 64-wide N tile maps to exactly one of {q,k,v} and one head, so routing
// is computed once per tile. Results scattered (bf16) into:
//   q : [B,H,S,64]   (row-major over head dim)          -> A operand (scores)
//   k : [B,H,S,64]   (col=key is head-dim-contiguous)   -> B operand (scores)
//   vT: [B,H,64,S]   (col=head-dim is key-contiguous)   -> B operand (P@V)
// ---------------------------------------------------------------------------
__global__ __launch_bounds__(128) void qkv_gemm_kernel(
    const uint16_t* __restrict__ xb, const uint16_t* __restrict__ WT,
    const float* __restrict__ bias, uint16_t* __restrict__ qb,
    uint16_t* __restrict__ kbuf, uint16_t* __restrict__ vT) {
    const int lane = threadIdx.x & 31;
    const int wave = threadIdx.x >> 5;
    const int tile = blockIdx.x * 4 + wave;
    const int NT = NQKV / 64;  // 36 macro-tiles along N
    const int mt = tile / NT, nt = tile % NT;
    const int m0 = mt * 32, n0 = nt * 64;
    const int l15 = lane & 15, lh = lane >> 4;

    const uint16_t* arow0 = xb + (size_t)(m0 + l15) * DIM + lh * 16;
    const uint16_t* arow1 = arow0 + (size_t)16 * DIM;
    const uint16_t* bcol = WT + (size_t)(n0 + l15) * DIM + lh * 16;

    v8f acc[2][4];
#pragma unroll
    for (int i = 0; i < 2; ++i)
#pragma unroll
        for (int j = 0; j < 4; ++j) acc[i][j] = zero8();

    for (int k0 = 0; k0 < DIM; k0 += 32) {
        const v16bf a0 = load_a_frag(arow0 + k0);
        const v16bf a1 = load_a_frag(arow1 + k0);
#pragma unroll
        for (int j = 0; j < 4; ++j) {
            const v16bf b = load_b_frag(bcol + (size_t)j * 16 * DIM + k0);
            acc[0][j] = wmma_bf16(a0, b, acc[0][j]);
            acc[1][j] = wmma_bf16(a1, b, acc[1][j]);
        }
    }

    // whole 64-wide tile lives in one segment + one head
    const int seg = n0 / DIM;      // 0=q, 1=k, 2=v
    const int cw0 = n0 % DIM;      // multiple of 64
    const int hh = cw0 / HD;

#pragma unroll
    for (int j = 0; j < 4; ++j) {
        const int col = n0 + j * 16 + l15;
        const int d = j * 16 + l15;        // head-dim index
        const float bv = bias[col];
#pragma unroll
        for (int i = 0; i < 2; ++i) {
#pragma unroll
            for (int r = 0; r < 8; ++r) {
                const int m = m0 + i * 16 + r + lh * 8;  // row in [0,4096)
                const int bi = m >> 11;                  // batch
                const int s = m & (SEQ - 1);             // seq pos
                const uint16_t hv = f2bf(acc[i][j][r] + bv);
                const size_t bh = (size_t)(bi * NHEADS + hh);
                if (seg == 0)      qb[(bh * SEQ + s) * HD + d] = hv;
                else if (seg == 1) kbuf[(bh * SEQ + s) * HD + d] = hv;
                else               vT[(bh * HD + d) * SEQ + s] = hv;
            }
        }
    }
}

// ---------------------------------------------------------------------------
// Kernel 3: causal flash attention. One wave (32 threads) per
// (batch, head, 16-query tile). Key blocks of 32.
// Scores: 4x wmma bf16 (K=64 split into two 32-chunks, two 16-col tiles).
// Online softmax in f32 registers; P converted to A-layout via 1KB LDS tile.
// P@V: 4x wmma bf16 into the 16x64 running accumulator.
// Next key block's K/V lines are prefetched (global_prefetch_b8) while the
// softmax VALU work runs. Output bf16 [B,S,768].
// ---------------------------------------------------------------------------
__global__ __launch_bounds__(32) void flash_attn_kernel(
    const uint16_t* __restrict__ qb, const uint16_t* __restrict__ kbuf,
    const uint16_t* __restrict__ vT, uint16_t* __restrict__ attnb) {
    const int lane = threadIdx.x & 31;
    const int qt = blockIdx.x & 127;              // S/16 tiles
    const int h = (blockIdx.x >> 7) % NHEADS;
    const int b = blockIdx.x / (128 * NHEADS);
    const size_t bh = (size_t)(b * NHEADS + h);
    const uint16_t* qp = qb + bh * SEQ * HD;
    const uint16_t* kp = kbuf + bh * SEQ * HD;
    const uint16_t* vp = vT + bh * HD * SEQ;

    const int l15 = lane & 15;
    const int lh = lane >> 4;

    // Q fragments for this query tile (head dim 64 = two K=32 chunks)
    const uint16_t* qrow = qp + (size_t)(qt * 16 + l15) * HD + lh * 16;
    const v16bf qf0 = load_a_frag(qrow);
    const v16bf qf1 = load_a_frag(qrow + 32);

    float mrow[8], lrow[8];
#pragma unroll
    for (int r = 0; r < 8; ++r) { mrow[r] = -1e30f; lrow[r] = 0.f; }
    v8f O0 = zero8(), O1 = zero8(), O2 = zero8(), O3 = zero8();

    __shared__ uint16_t Pt[16 * 32];

    const int nkb = (qt >> 1) + 1;                // causal: 32-key blocks
    for (int kb = 0; kb < nkb; ++kb) {
        const int kbase = kb * 32;

        // ---- scores: S = (Q K^T) * 1/sqrt(64), two 16-col tiles ----
        v8f s0 = zero8(), s1 = zero8();
        {
            const uint16_t* kc0 = kp + (size_t)(kbase + l15) * HD + lh * 16;
            const uint16_t* kc1 = kp + (size_t)(kbase + 16 + l15) * HD + lh * 16;
            s0 = wmma_bf16(qf0, load_b_frag(kc0), s0);
            s0 = wmma_bf16(qf1, load_b_frag(kc0 + 32), s0);
            s1 = wmma_bf16(qf0, load_b_frag(kc1), s1);
            s1 = wmma_bf16(qf1, load_b_frag(kc1 + 32), s1);
        }

        // prefetch next key block's K / V lines during softmax VALU work
        if (kb + 1 < nkb) {
            const int nb = kbase + 32;
            __builtin_prefetch(kp + (size_t)(nb + l15) * HD + lh * 16, 0, 1);
            __builtin_prefetch(kp + (size_t)(nb + 16 + l15) * HD + lh * 16, 0, 1);
            __builtin_prefetch(vp + (size_t)l15 * SEQ + nb + lh * 16, 0, 1);
            __builtin_prefetch(vp + (size_t)(32 + l15) * SEQ + nb + lh * 16, 0, 1);
        }

        // ---- online softmax per row register ----
#pragma unroll
        for (int r = 0; r < 8; ++r) {
            const int row = qt * 16 + r + lh * 8;
            const int c0 = kbase + l15;
            const int c1 = c0 + 16;
            float v0 = (c0 <= row) ? s0[r] * 0.125f : -1e30f;
            float v1 = (c1 <= row) ? s1[r] * 0.125f : -1e30f;
            float mx = fmaxf(v0, v1);
            mx = fmaxf(mx, __shfl_xor(mx, 1, 32));
            mx = fmaxf(mx, __shfl_xor(mx, 2, 32));
            mx = fmaxf(mx, __shfl_xor(mx, 4, 32));
            mx = fmaxf(mx, __shfl_xor(mx, 8, 32));
            const float mn = fmaxf(mrow[r], mx);
            const float alpha = __expf(mrow[r] - mn);
            mrow[r] = mn;
            const float p0 = __expf(v0 - mn);
            const float p1 = __expf(v1 - mn);
            float rs = p0 + p1;
            rs += __shfl_xor(rs, 1, 32);
            rs += __shfl_xor(rs, 2, 32);
            rs += __shfl_xor(rs, 4, 32);
            rs += __shfl_xor(rs, 8, 32);
            lrow[r] = lrow[r] * alpha + rs;
            O0[r] *= alpha; O1[r] *= alpha; O2[r] *= alpha; O3[r] *= alpha;
            // stash P (bf16) in LDS, row-major 16x32
            const int lr = r + lh * 8;
            Pt[lr * 32 + l15] = f2bf(p0);
            Pt[lr * 32 + 16 + l15] = f2bf(p1);
        }

        // ---- reload P in A-layout (C-layout -> A-layout transpose via LDS) --
        const v16bf pf = load_a_frag(&Pt[l15 * 32 + lh * 16]);

        // ---- P @ V: four 16-dim column tiles ----
        const uint16_t* vb = vp + (size_t)l15 * SEQ + kbase + lh * 16;
        O0 = wmma_bf16(pf, load_b_frag(vb + 0 * 16 * SEQ), O0);
        O1 = wmma_bf16(pf, load_b_frag(vb + 1 * 16 * SEQ), O1);
        O2 = wmma_bf16(pf, load_b_frag(vb + 2 * 16 * SEQ), O2);
        O3 = wmma_bf16(pf, load_b_frag(vb + 3 * 16 * SEQ), O3);
    }

    // ---- epilogue: O / l, write bf16 [B, S, 768] ----
#pragma unroll
    for (int r = 0; r < 8; ++r) {
        const float inv = 1.0f / lrow[r];
        const int s = qt * 16 + r + lh * 8;
        uint16_t* orow = attnb + ((size_t)b * SEQ + s) * DIM + h * HD;
        orow[0 * 16 + l15] = f2bf(O0[r] * inv);
        orow[1 * 16 + l15] = f2bf(O1[r] * inv);
        orow[2 * 16 + l15] = f2bf(O2[r] * inv);
        orow[3 * 16 + l15] = f2bf(O3[r] * inv);
    }
}

// ---------------------------------------------------------------------------
// Kernel 4: projection GEMM. out[4096,768] = attnb @ WprojT^T + b (f32 out)
// Same 32x64 macro-tile per wave as the QKV GEMM.
// ---------------------------------------------------------------------------
__global__ __launch_bounds__(128) void proj_gemm_kernel(
    const uint16_t* __restrict__ ab, const uint16_t* __restrict__ WT,
    const float* __restrict__ bias, float* __restrict__ out) {
    const int lane = threadIdx.x & 31;
    const int wave = threadIdx.x >> 5;
    const int tile = blockIdx.x * 4 + wave;
    const int NT = DIM / 64;  // 12 macro-tiles along N
    const int mt = tile / NT, nt = tile % NT;
    const int m0 = mt * 32, n0 = nt * 64;
    const int l15 = lane & 15, lh = lane >> 4;

    const uint16_t* arow0 = ab + (size_t)(m0 + l15) * DIM + lh * 16;
    const uint16_t* arow1 = arow0 + (size_t)16 * DIM;
    const uint16_t* bcol = WT + (size_t)(n0 + l15) * DIM + lh * 16;

    v8f acc[2][4];
#pragma unroll
    for (int i = 0; i < 2; ++i)
#pragma unroll
        for (int j = 0; j < 4; ++j) acc[i][j] = zero8();

    for (int k0 = 0; k0 < DIM; k0 += 32) {
        const v16bf a0 = load_a_frag(arow0 + k0);
        const v16bf a1 = load_a_frag(arow1 + k0);
#pragma unroll
        for (int j = 0; j < 4; ++j) {
            const v16bf b = load_b_frag(bcol + (size_t)j * 16 * DIM + k0);
            acc[0][j] = wmma_bf16(a0, b, acc[0][j]);
            acc[1][j] = wmma_bf16(a1, b, acc[1][j]);
        }
    }

#pragma unroll
    for (int j = 0; j < 4; ++j) {
        const int col = n0 + j * 16 + l15;
        const float bv = bias[col];
#pragma unroll
        for (int i = 0; i < 2; ++i) {
#pragma unroll
            for (int r = 0; r < 8; ++r) {
                const int m = m0 + i * 16 + r + lh * 8;
                out[(size_t)m * DIM + col] = acc[i][j][r] + bv;
            }
        }
    }
}

// ---------------------------------------------------------------------------
// Launch
// ---------------------------------------------------------------------------
extern "C" void kernel_launch(void* const* d_in, const int* in_sizes, int n_in,
                              void* d_out, int out_size, void* d_ws,
                              size_t ws_size, hipStream_t stream) {
    const float* x     = (const float*)d_in[0];  // [2,2048,768]
    const float* Wqkv  = (const float*)d_in[1];  // [768,2304]
    const float* bqkv  = (const float*)d_in[2];  // [2304]
    const float* Wproj = (const float*)d_in[3];  // [768,768]
    const float* bproj = (const float*)d_in[4];  // [768]
    float* out = (float*)d_out;

    char* ws = (char*)d_ws;
    const size_t NX = (size_t)MROWS * DIM;            // 3,145,728
    uint16_t* xb     = (uint16_t*)(ws);               // 6,291,456 B
    uint16_t* WqkvT  = (uint16_t*)(ws + 6291456);     // 3,538,944 B
    uint16_t* WprojT = (uint16_t*)(ws + 9830400);     // 1,179,648 B
    uint16_t* qb     = (uint16_t*)(ws + 11010048);    // 6,291,456 B
    uint16_t* kb     = (uint16_t*)(ws + 17301504);    // 6,291,456 B
    uint16_t* vT     = (uint16_t*)(ws + 23592960);    // 6,291,456 B
    uint16_t* attnb  = (uint16_t*)(ws + 29884416);    // 6,291,456 B
    // total workspace: 36,175,872 bytes

    // 1) precision conversion + weight transposes
    cvt_bf16_kernel<<<(int)(NX / 256), 256, 0, stream>>>(x, xb, (int)NX);
    cvt_transpose_kernel<<<(DIM * NQKV) / 256, 256, 0, stream>>>(
        Wqkv, WqkvT, DIM, NQKV);
    cvt_transpose_kernel<<<(DIM * DIM) / 256, 256, 0, stream>>>(
        Wproj, WprojT, DIM, DIM);

    // 2) QKV projection: (4096/32)*(2304/64) = 4608 macro-tiles, 4 waves/block
    qkv_gemm_kernel<<<4608 / 4, 128, 0, stream>>>(xb, WqkvT, bqkv, qb, kb, vT);

    // 3) flash attention: B*H*(S/16) = 3072 waves, one per block
    flash_attn_kernel<<<BATCH * NHEADS * (SEQ / 16), 32, 0, stream>>>(
        qb, kb, vT, attnb);

    // 4) output projection: (4096/32)*(768/64) = 1536 macro-tiles, 4 waves/blk
    proj_gemm_kernel<<<1536 / 4, 128, 0, stream>>>(attnb, WprojT, bproj, out);
}
